// VQAStandard_33028298506683
// MI455X (gfx1250) — matrix operand order repaired and checked by reading, not compiled, not run on hardware
//
#include <hip/hip_runtime.h>
#include <hip/hip_bf16.h>

// ---------------------------------------------------------------------------
// VQA 2-layer LSTM on gfx1250 (MI455X), bf16 WMMA (v_wmma_f32_16x16x32_bf16)
// B=128, T=26, E=300(pad 320), H=512, DIMG=4096, DOUT=1024
// ---------------------------------------------------------------------------

typedef __attribute__((ext_vector_type(16))) __bf16 v16bf;
typedef __attribute__((ext_vector_type(8)))  float  v8f;

#define BB   128
#define TT   26
#define EE   300
#define EPAD 320
#define HH   512
#define G4H  2048
#define DIMG 4096
#define DOUT 1024

union FragU { v16bf v; uint4 q[2]; };

// A fragment (16x32 bf16, M x K) from row-major bf16 src with leading dim ld.
// ISA layout: lanes 0-15: M=lane, K = K0+0..7 (v0-3) and K0+16..23 (v4-7)
//             lanes16-31: M=lane-16, K = K0+8..15 and K0+24..31
__device__ inline v16bf load_a_frag(const __bf16* src, int ld, int M0, int K0, int lane) {
    int row = M0 + (lane & 15);
    int kb  = K0 + ((lane >> 4) << 3);
    const __bf16* p = src + (size_t)row * ld + kb;
    FragU u;
    u.q[0] = *(const uint4*)p;          // K = kb .. kb+7
    u.q[1] = *(const uint4*)(p + 16);   // K = kb+16 .. kb+23
    return u.v;
}

// B fragment (32x16 bf16, K x N) from pre-packed buffer: fragment = (nt*KT+kt),
// per lane 16 contiguous bf16 (32 B).
__device__ inline v16bf load_b_frag(const __bf16* pk, int nt, int kt, int KT, int lane) {
    const __bf16* p = pk + (((size_t)(nt * KT + kt)) * 32 + lane) * 16;
    FragU u;
    u.q[0] = *(const uint4*)p;
    u.q[1] = *(const uint4*)(p + 8);
    return u.v;
}

__device__ inline v8f wmma_bf16(v16bf a, v16bf b, v8f c) {
    return __builtin_amdgcn_wmma_f32_16x16x32_bf16(false, a, false, b, (short)0, c, false, false);
}

__device__ inline float sigmoidf_(float x) { return 1.0f / (1.0f + __expf(-x)); }

// ---------------------------------------------------------------------------
// Setup: convert states / img_embed to working buffers
// ---------------------------------------------------------------------------
__global__ void vqa_init_states(const float* __restrict__ h01, const float* __restrict__ c01,
                                const float* __restrict__ h02, const float* __restrict__ c02,
                                const float* __restrict__ img,
                                __bf16* __restrict__ H1, float* __restrict__ C1,
                                __bf16* __restrict__ H2, float* __restrict__ C2,
                                __bf16* __restrict__ imgbf) {
    int i      = blockIdx.x * blockDim.x + threadIdx.x;
    int stride = gridDim.x * blockDim.x;
    for (int j = i; j < BB * HH; j += stride) {
        H1[j] = (__bf16)h01[j]; C1[j] = c01[j];
        H2[j] = (__bf16)h02[j]; C2[j] = c02[j];
    }
    for (int j = i; j < BB * DIMG; j += stride) imgbf[j] = (__bf16)img[j];
}

// ---------------------------------------------------------------------------
// Embedding gather -> time-major padded bf16 X [T*B, EPAD]
// ---------------------------------------------------------------------------
__global__ void vqa_gather_x(const int* __restrict__ desc,
                             const float* __restrict__ emb,
                             __bf16* __restrict__ X) {
    int row = blockIdx.x;            // row = t*B + b, 0..3327
    int t = row >> 7;                // /128
    int b = row & 127;
    int tok = desc[b * TT + t];
    const float* src = emb + (size_t)tok * EE;
    __bf16* dst = X + (size_t)row * EPAD;
    for (int j = threadIdx.x; j < EPAD; j += blockDim.x)
        dst[j] = (j < EE) ? (__bf16)src[j] : (__bf16)0.0f;
}

// ---------------------------------------------------------------------------
// Pack f32 weight W[N,K] (gemm uses B = W^T) into per-wave B-fragment layout.
// One wave per fragment.  B-frag: lanes 0-15: N=lane, K=K0+0..7,16..23 ; lanes16-31:
// N=lane-16, K=K0+8..15,24..31  (j = element index 0..15, pairs per VGPR)
// ---------------------------------------------------------------------------
__global__ void vqa_pack_w(const float* __restrict__ W, __bf16* __restrict__ pk,
                           int Ntiles, int Ktiles, int Ktrue, int ld) {
    int gw   = (blockIdx.x * blockDim.x + threadIdx.x) >> 5;
    int lane = threadIdx.x & 31;
    if (gw >= Ntiles * Ktiles) return;
    int nt = gw / Ktiles, kt = gw % Ktiles;
    int n  = nt * 16 + (lane & 15);
    int kb = kt * 32 + ((lane >> 4) << 3);
    __bf16 vals[16];
#pragma unroll
    for (int j = 0; j < 16; ++j) {
        int k = kb + (j < 8 ? j : j + 8);
        float f = (k < Ktrue) ? W[(size_t)n * ld + k] : 0.0f;
        vals[j] = (__bf16)f;
    }
    __bf16* dst = pk + ((size_t)gw * 32 + lane) * 16;
    *(uint4*)dst       = *(uint4*)&vals[0];
    *(uint4*)(dst + 8) = *(uint4*)&vals[8];
}

// ---------------------------------------------------------------------------
// G1pre = X @ W_ih1^T + (b_ih1 + b_hh1)   [3328, 2048] f32, K = 320 (padded)
// One wave per 16x16 tile.
// ---------------------------------------------------------------------------
__global__ void vqa_g1pre_gemm(const __bf16* __restrict__ X, const __bf16* __restrict__ Wpk,
                               const float* __restrict__ bih, const float* __restrict__ bhh,
                               float* __restrict__ G) {
    int gw   = (blockIdx.x * blockDim.x + threadIdx.x) >> 5;
    int lane = threadIdx.x & 31;
    const int MT = (TT * BB) / 16, NT = G4H / 16, KT = EPAD / 32;
    if (gw >= MT * NT) return;
    int mt = gw / NT, nt = gw % NT;
    int n15 = lane & 15, half = lane >> 4;
    float bias = bih[nt * 16 + n15] + bhh[nt * 16 + n15];
    v8f acc;
#pragma unroll
    for (int r = 0; r < 8; ++r) acc[r] = bias;
    for (int kt = 0; kt < KT; ++kt) {
        v16bf a = load_a_frag(X, EPAD, mt * 16, kt * 32, lane);
        v16bf b = load_b_frag(Wpk, nt, kt, KT, lane);
        acc = wmma_bf16(a, b, acc);
    }
#pragma unroll
    for (int r = 0; r < 8; ++r) {
        int row = mt * 16 + r + 8 * half;
        G[(size_t)row * G4H + nt * 16 + n15] = acc[r];
    }
}

// ---------------------------------------------------------------------------
// LSTM layer-1 step: gates = G1pre[t] + h1 @ W_hh1^T ; cell update.
// 256 waves: wave owns one (16-batch x 16-hidden) tile -> 4 gate accumulators.
// ---------------------------------------------------------------------------
__global__ void vqa_lstm_step1(const float* __restrict__ Gt, const __bf16* __restrict__ Whhpk,
                               const __bf16* __restrict__ h_in, __bf16* __restrict__ h_out,
                               float* __restrict__ C, __bf16* __restrict__ Acat) {
    int w    = (blockIdx.x * blockDim.x + threadIdx.x) >> 5;
    int lane = threadIdx.x & 31;
    if (w >= 256) return;
    int mt  = w >> 5;          // 0..7  (batch tile)
    int nth = w & 31;          // 0..31 (hidden-column tile)
    const int KT = HH / 32;    // 16
    int n15 = lane & 15, half = lane >> 4;

    v8f acc[4];
#pragma unroll
    for (int g = 0; g < 4; ++g) {
        int nt = g * 32 + nth;
#pragma unroll
        for (int r = 0; r < 8; ++r) {
            int row = mt * 16 + r + 8 * half;
            acc[g][r] = Gt[(size_t)row * G4H + nt * 16 + n15];
        }
    }
    for (int kt = 0; kt < KT; ++kt) {
        v16bf a = load_a_frag(h_in, HH, mt * 16, kt * 32, lane);
#pragma unroll
        for (int g = 0; g < 4; ++g) {
            v16bf b = load_b_frag(Whhpk, g * 32 + nth, kt, KT, lane);
            acc[g] = wmma_bf16(a, b, acc[g]);
        }
    }
#pragma unroll
    for (int r = 0; r < 8; ++r) {
        int row = mt * 16 + r + 8 * half;
        int col = nth * 16 + n15;
        float ig = sigmoidf_(acc[0][r]);
        float fg = sigmoidf_(acc[1][r]);
        float gg = tanhf(acc[2][r]);
        float og = sigmoidf_(acc[3][r]);
        float c  = C[(size_t)row * HH + col];
        float cn = fg * c + ig * gg;
        float hn = og * tanhf(cn);
        C[(size_t)row * HH + col]    = cn;
        h_out[(size_t)row * HH + col] = (__bf16)hn;
        Acat[(size_t)row * G4H + col]          = (__bf16)hn;  // h1 slice
        Acat[(size_t)row * G4H + HH + col]     = (__bf16)cn;  // c1 slice
    }
}

// ---------------------------------------------------------------------------
// LSTM layer-2 step: gates = h1 @ W_ih2^T + h2 @ W_hh2^T + b ; cell update.
// ---------------------------------------------------------------------------
__global__ void vqa_lstm_step2(const __bf16* __restrict__ Wihpk, const __bf16* __restrict__ Whhpk,
                               const float* __restrict__ bih, const float* __restrict__ bhh,
                               const __bf16* __restrict__ h1_new,
                               const __bf16* __restrict__ h2_in, __bf16* __restrict__ h2_out,
                               float* __restrict__ C, __bf16* __restrict__ Acat) {
    int w    = (blockIdx.x * blockDim.x + threadIdx.x) >> 5;
    int lane = threadIdx.x & 31;
    if (w >= 256) return;
    int mt  = w >> 5;
    int nth = w & 31;
    const int KT = HH / 32;
    int n15 = lane & 15, half = lane >> 4;

    v8f acc[4];
#pragma unroll
    for (int g = 0; g < 4; ++g) {
        float bias = bih[(g * 32 + nth) * 16 + n15] + bhh[(g * 32 + nth) * 16 + n15];
#pragma unroll
        for (int r = 0; r < 8; ++r) acc[g][r] = bias;
    }
    for (int kt = 0; kt < KT; ++kt) {                       // x-path: h1 (new)
        v16bf a = load_a_frag(h1_new, HH, mt * 16, kt * 32, lane);
#pragma unroll
        for (int g = 0; g < 4; ++g) {
            v16bf b = load_b_frag(Wihpk, g * 32 + nth, kt, KT, lane);
            acc[g] = wmma_bf16(a, b, acc[g]);
        }
    }
    for (int kt = 0; kt < KT; ++kt) {                       // recurrent path: h2 (old)
        v16bf a = load_a_frag(h2_in, HH, mt * 16, kt * 32, lane);
#pragma unroll
        for (int g = 0; g < 4; ++g) {
            v16bf b = load_b_frag(Whhpk, g * 32 + nth, kt, KT, lane);
            acc[g] = wmma_bf16(a, b, acc[g]);
        }
    }
#pragma unroll
    for (int r = 0; r < 8; ++r) {
        int row = mt * 16 + r + 8 * half;
        int col = nth * 16 + n15;
        float ig = sigmoidf_(acc[0][r]);
        float fg = sigmoidf_(acc[1][r]);
        float gg = tanhf(acc[2][r]);
        float og = sigmoidf_(acc[3][r]);
        float c  = C[(size_t)row * HH + col];
        float cn = fg * c + ig * gg;
        float hn = og * tanhf(cn);
        C[(size_t)row * HH + col]     = cn;
        h2_out[(size_t)row * HH + col] = (__bf16)hn;
        Acat[(size_t)row * G4H + 2 * HH + col] = (__bf16)hn;  // h2 slice
        Acat[(size_t)row * G4H + 3 * HH + col] = (__bf16)cn;  // c2 slice
    }
}

// ---------------------------------------------------------------------------
// Final fused: out = ([h1|c1|h2|c2] @ W_desc^T + b_desc) * (img @ W_img^T + b_img)
// One wave per 16x16 output tile; 8 x 64 = 512 waves.
// ---------------------------------------------------------------------------
__global__ void vqa_final(const __bf16* __restrict__ Acat, const __bf16* __restrict__ Wdpk,
                          const float* __restrict__ bd,
                          const __bf16* __restrict__ imgbf, const __bf16* __restrict__ Wipk,
                          const float* __restrict__ bi,
                          float* __restrict__ out) {
    int w    = (blockIdx.x * blockDim.x + threadIdx.x) >> 5;
    int lane = threadIdx.x & 31;
    if (w >= (BB / 16) * (DOUT / 16)) return;
    int mt = w >> 6;             // 0..7
    int nt = w & 63;             // 0..63
    int n15 = lane & 15, half = lane >> 4;

    const int KTd = G4H / 32;    // 64
    const int KTi = DIMG / 32;   // 128

    v8f accd, acci;
    float biasd = bd[nt * 16 + n15];
    float biasi = bi[nt * 16 + n15];
#pragma unroll
    for (int r = 0; r < 8; ++r) { accd[r] = biasd; acci[r] = biasi; }

    for (int kt = 0; kt < KTd; ++kt) {
        v16bf a = load_a_frag(Acat, G4H, mt * 16, kt * 32, lane);
        v16bf b = load_b_frag(Wdpk, nt, kt, KTd, lane);
        accd = wmma_bf16(a, b, accd);
    }
    for (int kt = 0; kt < KTi; ++kt) {
        v16bf a = load_a_frag(imgbf, DIMG, mt * 16, kt * 32, lane);
        v16bf b = load_b_frag(Wipk, nt, kt, KTi, lane);
        acci = wmma_bf16(a, b, acci);
    }
#pragma unroll
    for (int r = 0; r < 8; ++r) {
        int row = mt * 16 + r + 8 * half;
        out[(size_t)row * DOUT + nt * 16 + n15] = accd[r] * acci[r];
    }
}

// ---------------------------------------------------------------------------
extern "C" void kernel_launch(void* const* d_in, const int* in_sizes, int n_in,
                              void* d_out, int out_size, void* d_ws, size_t ws_size,
                              hipStream_t stream) {
    const int*   desc   = (const int*)  d_in[0];
    const float* img    = (const float*)d_in[1];
    const float* embm   = (const float*)d_in[2];
    const float* W_ih1  = (const float*)d_in[3];
    const float* W_hh1  = (const float*)d_in[4];
    const float* b_ih1  = (const float*)d_in[5];
    const float* b_hh1  = (const float*)d_in[6];
    const float* W_ih2  = (const float*)d_in[7];
    const float* W_hh2  = (const float*)d_in[8];
    const float* b_ih2  = (const float*)d_in[9];
    const float* b_hh2  = (const float*)d_in[10];
    const float* W_desc = (const float*)d_in[11];
    const float* b_desc = (const float*)d_in[12];
    const float* W_img  = (const float*)d_in[13];
    const float* b_img  = (const float*)d_in[14];
    const float* h01    = (const float*)d_in[15];
    const float* c01    = (const float*)d_in[16];
    const float* h02    = (const float*)d_in[17];
    const float* c02    = (const float*)d_in[18];

    // ---- workspace carve (bytes, 256B aligned) ----
    char* p = (char*)d_ws;
    auto carve = [&](size_t bytes) -> void* {
        void* r = (void*)p;
        p += (bytes + 255) & ~(size_t)255;
        return r;
    };
    __bf16* Xbf    = (__bf16*)carve((size_t)TT * BB * EPAD * 2);        // 2.1 MB
    float*  G1     = (float*) carve((size_t)TT * BB * G4H * 4);         // 27.3 MB
    __bf16* Wih1pk = (__bf16*)carve((size_t)128 * 10  * 512 * 2);       // 1.3 MB
    __bf16* Whh1pk = (__bf16*)carve((size_t)128 * 16  * 512 * 2);       // 2.1 MB
    __bf16* Wih2pk = (__bf16*)carve((size_t)128 * 16  * 512 * 2);
    __bf16* Whh2pk = (__bf16*)carve((size_t)128 * 16  * 512 * 2);
    __bf16* Wdpk   = (__bf16*)carve((size_t)64  * 64  * 512 * 2);       // 4.2 MB
    __bf16* Wipk   = (__bf16*)carve((size_t)64  * 128 * 512 * 2);       // 8.4 MB
    __bf16* H1     = (__bf16*)carve((size_t)2 * BB * HH * 2);           // ping-pong
    __bf16* H2     = (__bf16*)carve((size_t)2 * BB * HH * 2);
    float*  C1     = (float*) carve((size_t)BB * HH * 4);
    float*  C2     = (float*) carve((size_t)BB * HH * 4);
    __bf16* Acat   = (__bf16*)carve((size_t)BB * G4H * 2);
    __bf16* imgbf  = (__bf16*)carve((size_t)BB * DIMG * 2);
    (void)ws_size; (void)in_sizes; (void)n_in; (void)out_size;

    const int HSTRIDE = BB * HH; // elems per ping-pong buffer

    // ---- setup (independent of recurrence) ----
    vqa_init_states<<<256, 256, 0, stream>>>(h01, c01, h02, c02, img,
                                             H1, C1, H2, C2, imgbf);
    vqa_gather_x<<<TT * BB, 64, 0, stream>>>(desc, embm, Xbf);

    auto pack = [&](const float* W, __bf16* pk, int Nt, int Kt, int Ktrue, int ld) {
        int waves = Nt * Kt;
        vqa_pack_w<<<(waves * 32 + 255) / 256, 256, 0, stream>>>(W, pk, Nt, Kt, Ktrue, ld);
    };
    pack(W_ih1, Wih1pk, 128, 10,  EE,   EE);
    pack(W_hh1, Whh1pk, 128, 16,  HH,   HH);
    pack(W_ih2, Wih2pk, 128, 16,  HH,   HH);
    pack(W_hh2, Whh2pk, 128, 16,  HH,   HH);
    pack(W_desc, Wdpk,  64,  64,  G4H,  G4H);
    pack(W_img,  Wipk,  64,  128, DIMG, DIMG);

    // ---- hoisted input-path GEMM for all T ----
    {
        int waves = (TT * BB / 16) * (G4H / 16); // 26624
        vqa_g1pre_gemm<<<(waves * 32 + 255) / 256, 256, 0, stream>>>(
            Xbf, Wih1pk, b_ih1, b_hh1, G1);
    }

    // ---- sequential recurrence: 2 kernels per step ----
    for (int t = 0; t < TT; ++t) {
        int pp = t & 1;
        vqa_lstm_step1<<<32, 256, 0, stream>>>(
            G1 + (size_t)t * BB * G4H, Whh1pk,
            H1 + pp * HSTRIDE, H1 + (1 - pp) * HSTRIDE, C1, Acat);
        vqa_lstm_step2<<<32, 256, 0, stream>>>(
            Wih2pk, Whh2pk, b_ih2, b_hh2,
            H1 + (1 - pp) * HSTRIDE,
            H2 + pp * HSTRIDE, H2 + (1 - pp) * HSTRIDE, C2, Acat);
    }

    // ---- fused output GEMMs + elementwise product ----
    vqa_final<<<64, 256, 0, stream>>>(Acat, Wdpk, b_desc, imgbf, Wipk, b_img,
                                      (float*)d_out);
}
